// Head_42554535969178
// MI455X (gfx1250) — matrix operand order repaired and checked by reading, not compiled, not run on hardware
//
#include <hip/hip_runtime.h>
#include <stdint.h>

typedef __attribute__((ext_vector_type(16))) __bf16 v16bf;
typedef __attribute__((ext_vector_type(8)))  float  v8f;
typedef __attribute__((ext_vector_type(4)))  int    i32x4;

// addrspace-qualified vector types for the async-to-LDS builtin
typedef __attribute__((address_space(1))) i32x4 gas_i32x4;   // global
typedef __attribute__((address_space(3))) i32x4 las_i32x4;   // LDS

#define D_MODEL 768
#define HEAD    64
#define TSEQ    4096
#define NB      8
#define NKC     (D_MODEL / 32)   // 24 k-chunks

// ---- bf16 helpers: let the compiler use hardware cvt (RNE) ----------------
__device__ __forceinline__ uint32_t pack2(float lo, float hi) {
    union { __bf16 b[2]; uint32_t u; } cv;
    cv.b[0] = (__bf16)lo;
    cv.b[1] = (__bf16)hi;
    return cv.u;
}
__device__ __forceinline__ uint16_t bf16bits(float f) {
    union { __bf16 b; uint16_t u; } cv;
    cv.b = (__bf16)f;
    return cv.u;
}

union BfOp {
    uint32_t u[8];
    uint4    q[2];
    v16bf    v;
};

// ---- CDNA5 async global->LDS copy (16B per lane), guarded -----------------
__device__ __forceinline__ void async_copy16(const uint16_t* g, uint16_t* l) {
#if __has_builtin(__builtin_amdgcn_global_load_async_to_lds_b128)
    __builtin_amdgcn_global_load_async_to_lds_b128(
        (gas_i32x4*)g, (las_i32x4*)l, 0, 0);
#else
    *(uint4*)l = *(const uint4*)g;
#endif
}
__device__ __forceinline__ void async_wait0() {
#if __has_builtin(__builtin_amdgcn_global_load_async_to_lds_b128)
  #if __has_builtin(__builtin_amdgcn_s_wait_asynccnt)
    __builtin_amdgcn_s_wait_asynccnt(0);
  #else
    asm volatile("s_wait_asynccnt 0x0" ::: "memory");
  #endif
#endif
}

// ---------------------------------------------------------------------------
// Kernel 0: pre-swizzle W into bf16 WMMA-B operand layout.
// Wb[u32] layout: [mat 3][kc 24][n 4][lane 32][j 8]; element j packs
// K = kb+2j, kb+2j+1 where kb = kc*32 + 16*(lane>=16), col = n*16 + lane%16.
// ---------------------------------------------------------------------------
__global__ __launch_bounds__(256) void wswizzle(
    const float* __restrict__ Wq, const float* __restrict__ Wk,
    const float* __restrict__ Wv, uint32_t* __restrict__ Wb)
{
    int idx = blockIdx.x * 256 + threadIdx.x;        // < 3*24*4*256 = 73728
    int t = idx;
    const int j    = t & 7;   t >>= 3;
    const int lane = t & 31;  t >>= 5;
    const int n    = t & 3;   t >>= 2;
    const int kc   = t % NKC;
    const int mat  = t / NKC;

    const float* W = (mat == 0) ? Wq : (mat == 1) ? Wk : Wv;
    const int hi  = lane >> 4;
    const int lm  = lane & 15;
    const int kb  = kc * 32 + hi * 16;
    const int col = n * 16 + lm;
    const int k0  = kb + 2 * j;
    Wb[idx] = pack2(W[k0 * HEAD + col], W[(k0 + 1) * HEAD + col]);
}

// ---------------------------------------------------------------------------
// Kernel 1: Q/K/V projection.  x fp32 -> Q,K bf16 [B*T,64] (Q pre-scaled by
// 1/sqrt(768)), V transposed bf16 [B,64,T].  One wave per 16 output rows.
// B-operands are two b128 loads from the pre-swizzled Wb.
// ---------------------------------------------------------------------------
__global__ __launch_bounds__(128) void qkv_proj(
    const float* __restrict__ x, const uint32_t* __restrict__ Wb,
    uint16_t* __restrict__ Qs, uint16_t* __restrict__ Ks, uint16_t* __restrict__ Vt)
{
    const int lane = threadIdx.x & 31;
    const int wave = threadIdx.x >> 5;
    const int tile = blockIdx.x * 4 + wave;   // 2048 tiles of 16 rows
    const int t0   = tile * 16;               // global row = b*T + t
    const int hi   = lane >> 4;
    const int lm   = lane & 15;

    v8f acc[3][4];
#pragma unroll
    for (int m = 0; m < 3; m++)
#pragma unroll
        for (int n = 0; n < 4; n++) acc[m][n] = (v8f)0.f;

    const int arow = t0 + lm;
    const int aoff = hi ? 16 : 0;

    for (int kc = 0; kc < NKC; kc++) {
        // ---- A tile: x[arow, kc*32 + aoff + {0..15}] (contiguous k order)
        BfOp A;
        const float* xp = x + (size_t)arow * D_MODEL + kc * 32 + aoff;
        float4 f0 = *(const float4*)(xp + 0);
        float4 f1 = *(const float4*)(xp + 4);
        float4 f2 = *(const float4*)(xp + 8);
        float4 f3 = *(const float4*)(xp + 12);
        A.u[0] = pack2(f0.x, f0.y); A.u[1] = pack2(f0.z, f0.w);
        A.u[2] = pack2(f1.x, f1.y); A.u[3] = pack2(f1.z, f1.w);
        A.u[4] = pack2(f2.x, f2.y); A.u[5] = pack2(f2.z, f2.w);
        A.u[6] = pack2(f3.x, f3.y); A.u[7] = pack2(f3.z, f3.w);

#pragma unroll
        for (int m = 0; m < 3; m++) {
#pragma unroll
            for (int n = 0; n < 4; n++) {
                const uint32_t* wb = Wb + (((m * NKC + kc) * 4 + n) * 32 + lane) * 8;
                BfOp Bv;
                Bv.q[0] = *(const uint4*)(wb);
                Bv.q[1] = *(const uint4*)(wb + 4);
                acc[m][n] = __builtin_amdgcn_wmma_f32_16x16x32_bf16(
                    false, A.v, false, Bv.v, (short)0, acc[m][n], false, false);
            }
        }
    }

    const float scale = 0.03608439182435161f;   // 1/sqrt(768)
    const int b     = t0 / TSEQ;
    const int tloc0 = t0 - b * TSEQ;
#pragma unroll
    for (int n = 0; n < 4; n++) {
        const int h = n * 16 + lm;
#pragma unroll
        for (int r = 0; r < 8; r++) {
            const int trow = t0 + r + (hi ? 8 : 0);
            Qs[(size_t)trow * HEAD + h] = bf16bits(acc[0][n][r] * scale);
            Ks[(size_t)trow * HEAD + h] = bf16bits(acc[1][n][r]);
            const int tl = tloc0 + r + (hi ? 8 : 0);
            Vt[((size_t)b * HEAD + h) * TSEQ + tl] = bf16bits(acc[2][n][r]);
        }
    }
}

// ---------------------------------------------------------------------------
// Kernel 2: flash attention.  8 waves/block share K/V tiles staged in LDS by
// async global->LDS copies.  Each wave owns 16 q-rows x full head dim.
// S^T = K*Q^T so P^T registers are already in WMMA-A layout for P*V.
// ---------------------------------------------------------------------------
__global__ __launch_bounds__(256) void flash_attn(
    const uint16_t* __restrict__ Qs, const uint16_t* __restrict__ Ks,
    const uint16_t* __restrict__ Vt, const int* __restrict__ mask,
    float* __restrict__ out)
{
    __shared__ uint16_t ktile[32 * HEAD];   // [key 32][h 64]   4 KB
    __shared__ uint16_t vtile[HEAD * 32];   // [h 64][key 32]   4 KB

    const int tid  = threadIdx.x;
    const int lane = tid & 31;
    const int wave = tid >> 5;
    const int tile = blockIdx.x * 8 + wave;
    const int t0   = tile * 16;              // global q-row base
    const int b    = t0 / TSEQ;
    const int hi   = lane >> 4;
    const int lm   = lane & 15;

    // Q^T B-operands: element e <-> h = hi*16 + e (+32 for bq1)
    BfOp bq0, bq1;
    {
        const uint16_t* qp = Qs + (size_t)(t0 + lm) * HEAD + (hi ? 16 : 0);
        bq0.q[0] = *(const uint4*)(qp);
        bq0.q[1] = *(const uint4*)(qp + 8);
        bq1.q[0] = *(const uint4*)(qp + 32);
        bq1.q[1] = *(const uint4*)(qp + 40);
    }

    v8f acc[4];
#pragma unroll
    for (int n = 0; n < 4; n++) acc[n] = (v8f)0.f;
    float m_i = -3.0e38f, l_i = 0.f;

    const int vh = tid >> 2;                 // staging coords for V tile
    const int vj = (tid & 3) * 8;

    for (int kt = 0; kt < TSEQ; kt += 32) {
        // ---- stage K tile (contiguous 4KB) and V^T tile (64 rows x 64B)
        async_copy16(Ks + ((size_t)b * TSEQ + kt) * HEAD + tid * 8, ktile + tid * 8);
        async_copy16(Vt + ((size_t)b * HEAD + vh) * TSEQ + kt + vj, vtile + vh * 32 + vj);

        const int   mv  = mask[(size_t)b * TSEQ + kt + lane];
        const float neg = mv ? 0.f : -1.0e30f;

        async_wait0();
        __syncthreads();

        // ---- S^T = K * Q^T (two 16-key subtiles, K-dim 64 = 2 WMMAs each)
        v8f s[2];
#pragma unroll
        for (int sub = 0; sub < 2; sub++) {
            const uint16_t* kp = ktile + (sub * 16 + lm) * HEAD + (hi ? 16 : 0);
            BfOp a0, a1;
            a0.q[0] = *(const uint4*)(kp);        // h off..off+7
            a0.q[1] = *(const uint4*)(kp + 8);    // h off+8..+15
            a1.q[0] = *(const uint4*)(kp + 32);
            a1.q[1] = *(const uint4*)(kp + 40);
            v8f c = (v8f)0.f;
            c = __builtin_amdgcn_wmma_f32_16x16x32_bf16(false, a0.v, false, bq0.v,
                                                        (short)0, c, false, false);
            c = __builtin_amdgcn_wmma_f32_16x16x32_bf16(false, a1.v, false, bq1.v,
                                                        (short)0, c, false, false);
            s[sub] = c;
        }

        // ---- mask bias: element (sub,r) has key offset sub*16 + r + 8*hi
#pragma unroll
        for (int sub = 0; sub < 2; sub++)
#pragma unroll
            for (int r = 0; r < 8; r++)
                s[sub][r] += __shfl(neg, sub * 16 + r + (hi ? 8 : 0), 32);

        // ---- online softmax over keys (q = lane%16; combine lane halves)
        float tmax = s[0][0];
#pragma unroll
        for (int r = 1; r < 8; r++) tmax = fmaxf(tmax, s[0][r]);
#pragma unroll
        for (int r = 0; r < 8; r++) tmax = fmaxf(tmax, s[1][r]);
        tmax = fmaxf(tmax, __shfl_xor(tmax, 16, 32));
        const float m_new = fmaxf(m_i, tmax);
        const float corr  = __expf(m_i - m_new);

        float rsum = 0.f;
        BfOp P;                               // P^T -> WMMA-A layout
#pragma unroll
        for (int sub = 0; sub < 2; sub++)
#pragma unroll
            for (int r = 0; r < 8; r += 2) {
                float p0 = __expf(s[sub][r]     - m_new);
                float p1 = __expf(s[sub][r + 1] - m_new);
                rsum += p0 + p1;
                P.u[sub * 4 + r / 2] = pack2(p0, p1);
            }
        rsum += __shfl_xor(rsum, 16, 32);
        l_i = l_i * corr + rsum;
        m_i = m_new;

        // ---- rescale accumulators: acc row r is q = r + 8*hi
        float cb[8];
#pragma unroll
        for (int r = 0; r < 8; r++) cb[r] = __shfl(corr, r + (hi ? 8 : 0), 32);
#pragma unroll
        for (int n = 0; n < 4; n++)
#pragma unroll
            for (int r = 0; r < 8; r++) acc[n][r] *= cb[r];

        // ---- acc += P * V (V B-operands from LDS, two 16B reads at keys 8*hi and 16+8*hi)
#pragma unroll
        for (int n = 0; n < 4; n++) {
            const uint16_t* vp = vtile + (n * 16 + lm) * 32 + (hi ? 8 : 0);
            BfOp Bv;
            Bv.q[0] = *(const uint4*)(vp);
            Bv.q[1] = *(const uint4*)(vp + 16);
            acc[n] = __builtin_amdgcn_wmma_f32_16x16x32_bf16(
                false, P.v, false, Bv.v, (short)0, acc[n], false, false);
        }

        __syncthreads();   // protect tiles before next stage
    }

    // ---- normalize and store fp32 output
    float lb[8];
#pragma unroll
    for (int r = 0; r < 8; r++)
        lb[r] = 1.0f / __shfl(l_i, r + (hi ? 8 : 0), 32);
#pragma unroll
    for (int n = 0; n < 4; n++) {
        const int h = n * 16 + lm;
#pragma unroll
        for (int r = 0; r < 8; r++) {
            const int trow = t0 + r + (hi ? 8 : 0);
            out[(size_t)trow * HEAD + h] = acc[n][r] * lb[r];
        }
    }
}

extern "C" void kernel_launch(void* const* d_in, const int* in_sizes, int n_in,
                              void* d_out, int out_size, void* d_ws, size_t ws_size,
                              hipStream_t stream) {
    const float* x    = (const float*)d_in[0];
    const int*   mask = (const int*)d_in[1];
    const float* Wq   = (const float*)d_in[2];
    const float* Wk   = (const float*)d_in[3];
    const float* Wv   = (const float*)d_in[4];
    float* out = (float*)d_out;

    uint16_t* Qs = (uint16_t*)d_ws;                        // 4 MB
    uint16_t* Ks = Qs + (size_t)NB * TSEQ * HEAD;          // 4 MB
    uint16_t* Vt = Ks + (size_t)NB * TSEQ * HEAD;          // 4 MB
    uint32_t* Wb = (uint32_t*)(Vt + (size_t)NB * TSEQ * HEAD);   // 288 KB

    wswizzle <<<dim3(288), dim3(256), 0, stream>>>(Wq, Wk, Wv, Wb);
    qkv_proj <<<dim3(NB * TSEQ / 64), dim3(128), 0, stream>>>(x, Wb, Qs, Ks, Vt);
    flash_attn<<<dim3(NB * TSEQ / 128), dim3(256), 0, stream>>>(Qs, Ks, Vt, mask, out);
}
